// SurfEdgeDecoder_40999757808028
// MI455X (gfx1250) — matrix operand order repaired and checked by reading, not compiled, not run on hardware
//
#include <hip/hip_runtime.h>

typedef __attribute__((ext_vector_type(16))) __bf16 v16bf;
typedef __attribute__((ext_vector_type(8)))  float  v8f;

#define IN_CH   128
#define TWO_C   256
#define HIDDEN  256
#define ROWP    264              // padded LDS row stride (bf16 elems): 528B = 132 dwords -> bank spread
#define EPB     64               // edges per block (4 WMMA M-tiles)
#define MT      4

union FragU {
  v16bf v;
  struct { uint4 lo; uint4 hi; } u;
};

__device__ __forceinline__ unsigned short f2bf(float f) {
  unsigned int u = __float_as_uint(f);
  u = u + 0x7FFFu + ((u >> 16) & 1u);    // round-to-nearest-even
  return (unsigned short)(u >> 16);
}

__global__ void w1_to_bf16(const float* __restrict__ w1, unsigned short* __restrict__ w1bf) {
  int i = blockIdx.x * blockDim.x + threadIdx.x;
  if (i < HIDDEN * TWO_C) w1bf[i] = f2bf(w1[i]);
}

__global__ __launch_bounds__(32)
void edge_decoder_kernel(const float* __restrict__ latent,
                         const unsigned short* __restrict__ w1bf,
                         const float* __restrict__ b1,
                         const float* __restrict__ w2,
                         const float* __restrict__ b2,
                         const long long* __restrict__ eidx,
                         float* __restrict__ out,
                         int E) {
  __shared__ unsigned short Abuf[EPB * ROWP];   // 64 x 264 bf16 = 33,792 B

  const int lane = threadIdx.x;
  const int base = blockIdx.x * EPB;
  const long long Em1 = (long long)E - 1;

  // Each lane caches indices for 2 edges; broadcast via shfl in the gather loop.
  long long eA = (long long)(base + lane);        if (eA > Em1) eA = Em1;
  long long eB = (long long)(base + 32 + lane);   if (eB > Em1) eB = Em1;
  const int sA = (int)eidx[eA], dA = (int)eidx[(long long)E + eA];
  const int sB = (int)eidx[eB], dB = (int)eidx[(long long)E + eB];

  // ---- Gather 64 edges' [src|dst] rows, fp32 -> bf16, into LDS ----
  // 128 floats per half-row / 32 lanes = one float4 per lane per half.
  #pragma unroll 4
  for (int e = 0; e < EPB; ++e) {
    int s = __shfl((e < 32) ? sA : sB, e & 31);
    int d = __shfl((e < 32) ? dA : dB, e & 31);
    float4 vs = ((const float4*)(latent + (size_t)s * IN_CH))[lane];
    float4 vd = ((const float4*)(latent + (size_t)d * IN_CH))[lane];
    unsigned short* row = &Abuf[e * ROWP];
    ushort4 ps = { f2bf(vs.x), f2bf(vs.y), f2bf(vs.z), f2bf(vs.w) };
    ushort4 pd = { f2bf(vd.x), f2bf(vd.y), f2bf(vd.z), f2bf(vd.w) };
    ((ushort4*)row)[lane]            = ps;   // src channels -> cols [0,128)
    ((ushort4*)(row + IN_CH))[lane]  = pd;   // dst channels -> cols [128,256)
  }
  __syncthreads();

  const int m     = lane & 15;       // A row within M-tile / B column within N-tile
  const int hi    = lane >> 4;       // half-wave select
  const int koffA = hi * 8;          // A layout: lanes 16-31 hold K+8 / K+24
  const int koffB = hi * 8;          // B layout: lanes 16-31 hold K+8 / K+24

  float logit[MT][8];
  #pragma unroll
  for (int mt = 0; mt < MT; ++mt)
    #pragma unroll
    for (int r = 0; r < 8; ++r) logit[mt][r] = 0.0f;

  const float b2v = b2[0];

  // ---- GEMM1 (WMMA bf16) fused with bias+ReLU and layer-2 dot product ----
  #pragma unroll 1
  for (int nt = 0; nt < 16; ++nt) {
    const int n = nt * 16 + m;                        // hidden-unit index for this lane
    const uint4* w1row = (const uint4*)(w1bf + (size_t)n * TWO_C);

    v8f acc[MT];
    #pragma unroll
    for (int mt = 0; mt < MT; ++mt) acc[mt] = (v8f){0.f,0.f,0.f,0.f,0.f,0.f,0.f,0.f};

    #pragma unroll
    for (int kf = 0; kf < 8; ++kf) {
      const int kb = kf * 32;
      FragU bf;                                        // B: W1^T tile (lane = column n)
      bf.u.lo = w1row[(kb + koffB) >> 3];
      bf.u.hi = w1row[((kb + koffB) >> 3) + 2];
      #pragma unroll
      for (int mt = 0; mt < MT; ++mt) {
        const unsigned short* arow = &Abuf[(mt * 16 + m) * ROWP + kb + koffA];
        FragU af;
        af.u.lo = *(const uint4*)(arow);               // K kb+koffA   .. +7
        af.u.hi = *(const uint4*)(arow + 16);          // K kb+16+koffA.. +7
        acc[mt] = __builtin_amdgcn_wmma_f32_16x16x32_bf16(
            false, af.v, false, bf.v, (short)0, acc[mt], false, false);
      }
    }

    const float b1v = b1[n];
    const float w2v = w2[n];
    #pragma unroll
    for (int mt = 0; mt < MT; ++mt)
      #pragma unroll
      for (int r = 0; r < 8; ++r) {
        float h = acc[mt][r] + b1v;
        logit[mt][r] += fmaxf(h, 0.0f) * w2v;          // fused layer-2 partial dot
      }
  }

  // ---- Reduce over the 16 hidden-unit lanes of each half-wave ----
  #pragma unroll
  for (int mt = 0; mt < MT; ++mt)
    #pragma unroll
    for (int r = 0; r < 8; ++r) {
      float v = logit[mt][r];
      v += __shfl_xor(v, 1);
      v += __shfl_xor(v, 2);
      v += __shfl_xor(v, 4);
      v += __shfl_xor(v, 8);
      logit[mt][r] = v;
    }

  // lanes 0 and 16 hold full sums: lane0 -> rows M=0..7, lane16 -> rows M=8..15
  if ((lane & 15) == 0) {
    #pragma unroll
    for (int mt = 0; mt < MT; ++mt) {
      const int ebase = base + mt * 16 + hi * 8;
      #pragma unroll
      for (int r = 0; r < 8; ++r) {
        const int e = ebase + r;
        if (e < E) {
          float x = logit[mt][r] + b2v;
          out[e] = 1.0f / (1.0f + __expf(-x));
        }
      }
    }
  }
}

extern "C" void kernel_launch(void* const* d_in, const int* in_sizes, int n_in,
                              void* d_out, int out_size, void* d_ws, size_t ws_size,
                              hipStream_t stream) {
  const float*     latent = (const float*)d_in[0];
  const float*     W1     = (const float*)d_in[1];
  const float*     b1     = (const float*)d_in[2];
  const float*     W2     = (const float*)d_in[3];
  const float*     b2     = (const float*)d_in[4];
  const long long* eidx   = (const long long*)d_in[5];
  const int E = in_sizes[5] / 2;

  unsigned short* w1bf = (unsigned short*)d_ws;   // 128 KB bf16 copy of W1, rebuilt every call

  w1_to_bf16<<<(HIDDEN * TWO_C + 255) / 256, 256, 0, stream>>>(W1, w1bf);

  const int nblocks = (E + EPB - 1) / EPB;        // 7813 single-wave blocks
  edge_decoder_kernel<<<nblocks, 32, 0, stream>>>(latent, w1bf, b1, W2, b2, eidx,
                                                  (float*)d_out, E);
}
